// MFAGLALayer_62302795596499
// MI455X (gfx1250) — compile-verified, matrix operand order checked by reading, not run on hardware
//
#include <hip/hip_runtime.h>
#include <hip/hip_bf16.h>
#include <math.h>

// MFA-GLA layer for MI455X (gfx1250, wave32, WMMA).
// B=8 T=2048 D=1024 H=8 C=64 DV=128
#define Bb 8
#define Tt 2048
#define Dd 1024
#define Hh 8
#define Cc 64
#define DVv 128
#define BT (Bb * Tt)
#define NPROJ 704          // q(64) + k(64) + v(64) + alpha(512)
#define NPROJ_PAD 768      // padded to 48 N-tiles so 8 waves x 6 tiles uniform

typedef __attribute__((ext_vector_type(16))) __bf16 v16bf;
typedef __attribute__((ext_vector_type(8)))  float  v8f;

// ---------------------------------------------------------------------------
// WMMA fragment layout helpers (16x16x32 bf16, wave32).
// A (16x32): lane l holds row M=l%16; element j (0..15) maps to
//   K = (j<8 ? j : j+8) + (l<16 ? 0 : 8). K-pairs (2i,2i+1) are contiguous.
// B (32x16): lane l holds column N=l%16 with the same K mapping (baked into
//   pre-converted bf16 "panel" blocks of 512 elements -> one contiguous 32B
//   chunk per lane).
// C/D (16x16 f32): lane l, vgpr g -> M = g + (l<16?0:8), N = l%16.
// ---------------------------------------------------------------------------

__device__ __host__ __forceinline__ int kmap(int lane, int j) {
  return ((j < 8) ? j : (j + 8)) + ((lane < 16) ? 0 : 8);
}

union Frag {
  v16bf v;
  unsigned u[8];
  __uint128_t o[2];
};

__device__ __forceinline__ v16bf load_a_lds(const __bf16* base, int ldk, int k0, int lane) {
  Frag r;
  const unsigned* p =
      (const unsigned*)(base + (lane & 15) * ldk + k0 + ((lane < 16) ? 0 : 8));
  r.u[0] = p[0]; r.u[1] = p[1]; r.u[2] = p[2]; r.u[3] = p[3];
  r.u[4] = p[8]; r.u[5] = p[9]; r.u[6] = p[10]; r.u[7] = p[11];
  return r.v;
}

__device__ __forceinline__ v16bf load_b_panel(const __bf16* panel, int blk, int lane) {
  Frag r;
  const __uint128_t* p = (const __uint128_t*)(panel + (size_t)blk * 512 + lane * 16);
  r.o[0] = p[0];
  r.o[1] = p[1];
  return r.v;
}

__device__ __forceinline__ v8f wmma_bf16(v16bf a, v16bf b, v8f c) {
  return __builtin_amdgcn_wmma_f32_16x16x32_bf16(false, a, false, b, (short)0, c,
                                                 false, false);
}

// ---------------------------------------------------------------------------
// CDNA5 async global->LDS copies (ASYNCcnt-tracked, no VGPR roundtrip).
// ---------------------------------------------------------------------------

__device__ __forceinline__ void async_load_b128(unsigned lds_off, const void* gaddr) {
  asm volatile("global_load_async_to_lds_b128 %0, %1, off"
               :
               : "v"(lds_off), "v"((unsigned long long)(size_t)gaddr)
               : "memory");
}

__device__ __forceinline__ void async_load_b32(unsigned lds_off, const void* gaddr) {
  asm volatile("global_load_async_to_lds_b32 %0, %1, off"
               :
               : "v"(lds_off), "v"((unsigned long long)(size_t)gaddr)
               : "memory");
}

__device__ __forceinline__ void async_wait0() {
  asm volatile("s_wait_asynccnt 0x0" ::: "memory");
}

__device__ __forceinline__ unsigned lds_off(const void* p) {
  return (unsigned)(size_t)p;
}

// ---------------------------------------------------------------------------
// Weight panelization: f32 -> bf16 fragment panels.
// nk variant: W is (N,K) row-major, used as  y @ W^T  (B[k][n] = W[n][k]).
//   Nsrc < gridN*16 rows are zero-padded.
// kn variant: W is (K,N) row-major, used as  y @ W     (B[k][n] = W[k][n]).
// ---------------------------------------------------------------------------

__global__ void __launch_bounds__(32) k_panel_nk(const float* __restrict__ W,
                                                 __bf16* __restrict__ dst, int Nsrc,
                                                 int K) {
  int kSteps = K >> 5;
  int blk = blockIdx.x;
  int nt = blk / kSteps, ks = blk % kSteps;
  int lane = threadIdx.x;
  __bf16* o = dst + (size_t)blk * 512 + lane * 16;
  int n = nt * 16 + (lane & 15);
  if (n < Nsrc) {
    for (int j = 0; j < 16; ++j)
      o[j] = (__bf16)W[(size_t)n * K + ks * 32 + kmap(lane, j)];
  } else {
    for (int j = 0; j < 16; ++j) o[j] = (__bf16)0.f;
  }
}

__global__ void __launch_bounds__(32) k_panel_kn(const float* __restrict__ W,
                                                 __bf16* __restrict__ dst, int N, int K) {
  int kSteps = K >> 5;
  int blk = blockIdx.x;
  int nt = blk / kSteps, ks = blk % kSteps;
  int lane = threadIdx.x;
  __bf16* o = dst + (size_t)blk * 512 + lane * 16;
  int n = nt * 16 + (lane & 15);
  for (int j = 0; j < 16; ++j)
    o[j] = (__bf16)W[(size_t)(ks * 32 + kmap(lane, j)) * N + n];
}

// ---------------------------------------------------------------------------
// Kernel 1: LayerNorm + fused combined projection (channels:
// [q_lat 64 | k_lat 64 | v_lat 64 | alpha 512 | zero-pad 64]) via WMMA over
// K=1024. One block = 16 rows; each of 8 waves owns exactly 6 N-tiles
// (uniform — no exec-masked WMMA). A-fragment loaded once per k-step; all 6
// B-fragments preloaded as a clause before the WMMA chain.
// ---------------------------------------------------------------------------

__global__ void __launch_bounds__(256) k_ln_proj(
    const float* __restrict__ x, const float* __restrict__ gamma,
    const float* __restrict__ beta, const float* __restrict__ wa_b,
    const __bf16* __restrict__ combP, __bf16* __restrict__ qlat,
    __bf16* __restrict__ vlat, float* __restrict__ kbuf,
    float* __restrict__ alpha) {
  __shared__ __attribute__((aligned(16))) __bf16 sxn[16 * Dd];
  __shared__ float ssum[256], ssq[256];
  __shared__ float smu[16], srs[16];

  int tid = threadIdx.x;
  int tile = blockIdx.x;
  int r = tid >> 4, seg = tid & 15;

  const float* xrow = x + (size_t)(tile * 16 + r) * Dd + seg * 64;
  float s = 0.f, s2 = 0.f;
  for (int i = 0; i < 64; ++i) {
    float v = xrow[i];
    s += v;
    s2 += v * v;
  }
  ssum[tid] = s;
  ssq[tid] = s2;
  __syncthreads();
  if (seg == 0) {
    float ts = 0.f, t2 = 0.f;
    for (int i = 0; i < 16; ++i) {
      ts += ssum[(r << 4) + i];
      t2 += ssq[(r << 4) + i];
    }
    float mu = ts * (1.f / Dd);
    float var = t2 * (1.f / Dd) - mu * mu;
    smu[r] = mu;
    srs[r] = rsqrtf(var + 1e-5f);
  }
  __syncthreads();
  {
    float mu = smu[r], rs = srs[r];
    for (int i = 0; i < 64; ++i) {
      int c = seg * 64 + i;
      float xn = (xrow[i] - mu) * rs * gamma[c] + beta[c];
      sxn[r * Dd + c] = (__bf16)xn;
    }
  }
  __syncthreads();

  // Scalar wave id -> SALU addressing, uniform branches, no exec-masked WMMA.
  int wave = __builtin_amdgcn_readfirstlane(tid >> 5);
  int lane = tid & 31;

  v8f zero = {};
  v8f acc[6];
#pragma unroll
  for (int j = 0; j < 6; ++j) acc[j] = zero;

  for (int ks = 0; ks < 32; ++ks) {
    v16bf a = load_a_lds(sxn, Dd, ks * 32, lane);
    if (ks + 1 < 32)
      __builtin_prefetch(combP + (size_t)(wave * 32 + ks + 1) * 512, 0, 1);
    v16bf bfrag[6];
#pragma unroll
    for (int j = 0; j < 6; ++j)
      bfrag[j] = load_b_panel(combP, (wave + 8 * j) * 32 + ks, lane);
#pragma unroll
    for (int j = 0; j < 6; ++j) acc[j] = wmma_bf16(a, bfrag[j], acc[j]);
  }

  for (int j = 0; j < 6; ++j) {
    int nt = wave + 8 * j;
    int n = nt * 16 + (lane & 15);
    if (n >= NPROJ) continue;  // zero-padded tiles, store side only
    for (int g = 0; g < 8; ++g) {
      int m = g + ((lane < 16) ? 0 : 8);
      int bt = tile * 16 + m;
      float val = acc[j][g];
      if (n < 64) {
        qlat[(size_t)bt * Cc + n] = (__bf16)val;
      } else if (n < 128) {
        kbuf[(size_t)bt * Cc + (n - 64)] = val;  // normalized later
      } else if (n < 192) {
        vlat[(size_t)bt * Cc + (n - 128)] = (__bf16)val;
      } else {
        int hc = n - 192;
        int h = hc >> 6, c = hc & 63;
        int b = bt >> 11, t = bt & (Tt - 1);
        float z = val + wa_b[hc];
        alpha[(((size_t)b * Hh + h) * Tt + t) * Cc + c] = 1.f / (1.f + __expf(-z));
      }
    }
  }
}

// ---------------------------------------------------------------------------
// Kernel 2: k row-normalize: k / max(||k||_2, 1e-12). One block per row.
// ---------------------------------------------------------------------------

__global__ void __launch_bounds__(64) k_knorm(float* __restrict__ kbuf) {
  __shared__ float sr[64];
  int row = blockIdx.x, c = threadIdx.x;
  float v = kbuf[(size_t)row * Cc + c];
  sr[c] = v * v;
  __syncthreads();
  for (int sft = 32; sft > 0; sft >>= 1) {
    if (c < sft) sr[c] += sr[c + sft];
    __syncthreads();
  }
  float scale = 1.f / fmaxf(sqrtf(sr[0]), 1e-12f);
  kbuf[(size_t)row * Cc + c] = v * scale;
}

// ---------------------------------------------------------------------------
// Kernel 3: per-head latent rotation q = q_lat @ Qc[h] and value up-proj
// v = v_lat @ Vc[h]. One wave per (row-tile, head). Tiles staged via async
// global->LDS b128 copies; A fragment shared across N-tiles per k-step.
// ---------------------------------------------------------------------------

__global__ void __launch_bounds__(32) k_rotate(
    const __bf16* __restrict__ qlat, const __bf16* __restrict__ vlat,
    const __bf16* __restrict__ qcP, const __bf16* __restrict__ vcP,
    float* __restrict__ q, float* __restrict__ v) {
  __shared__ __attribute__((aligned(16))) __bf16 slq[16 * Cc];
  __shared__ __attribute__((aligned(16))) __bf16 slv[16 * Cc];
  int tile = blockIdx.x, h = blockIdx.y;
  int lane = threadIdx.x;

  // 16x64 bf16 tile = 2 KB = 128 x b128 per matrix.
  const __uint128_t* qsrc = (const __uint128_t*)(qlat + (size_t)tile * 16 * Cc);
  const __uint128_t* vsrc = (const __uint128_t*)(vlat + (size_t)tile * 16 * Cc);
  unsigned bq = lds_off(slq), bv = lds_off(slv);
  for (int i = lane; i < 128; i += 32) {
    async_load_b128(bq + i * 16, qsrc + i);
    async_load_b128(bv + i * 16, vsrc + i);
  }
  async_wait0();
  __syncthreads();

  const __bf16* qp = qcP + (size_t)h * (4 * 2 * 512);
  const __bf16* vp = vcP + (size_t)h * (8 * 2 * 512);

  v8f zero = {};
  v8f aq[4], av[8];
#pragma unroll
  for (int j = 0; j < 4; ++j) aq[j] = zero;
#pragma unroll
  for (int j = 0; j < 8; ++j) av[j] = zero;

#pragma unroll
  for (int ks = 0; ks < 2; ++ks) {
    v16bf a1 = load_a_lds(slq, Cc, ks * 32, lane);
#pragma unroll
    for (int nt = 0; nt < 4; ++nt)
      aq[nt] = wmma_bf16(a1, load_b_panel(qp, nt * 2 + ks, lane), aq[nt]);
    v16bf a2 = load_a_lds(slv, Cc, ks * 32, lane);
#pragma unroll
    for (int nt = 0; nt < 8; ++nt)
      av[nt] = wmma_bf16(a2, load_b_panel(vp, nt * 2 + ks, lane), av[nt]);
  }

  for (int nt = 0; nt < 4; ++nt) {
    int n = nt * 16 + (lane & 15);
    for (int g = 0; g < 8; ++g) {
      int m = g + ((lane < 16) ? 0 : 8);
      int bt = tile * 16 + m;
      int b = bt >> 11, t = bt & (Tt - 1);
      q[(((size_t)b * Hh + h) * Tt + t) * Cc + n] = aq[nt][g];
    }
  }
  for (int nt = 0; nt < 8; ++nt) {
    int n = nt * 16 + (lane & 15);
    for (int g = 0; g < 8; ++g) {
      int m = g + ((lane < 16) ? 0 : 8);
      int bt = tile * 16 + m;
      int b = bt >> 11, t = bt & (Tt - 1);
      v[(((size_t)b * Hh + h) * Tt + t) * DVv + n] = av[nt][g];
    }
  }
}

// ---------------------------------------------------------------------------
// Kernel 4: gated scan. One block per (b,h); S (64x128 f32) lives entirely in
// registers (32 per thread). Per-step operands broadcast to LDS with async
// b32 copies (ASYNCcnt), then consumed by all 256 threads.
//   S = diag(a_t) S + kn_t v_t^T ;  o_t = q_t^T S
// ---------------------------------------------------------------------------

__global__ void __launch_bounds__(256) k_scan(
    const float* __restrict__ q, const float* __restrict__ kn,
    const float* __restrict__ v, const float* __restrict__ alpha,
    __bf16* __restrict__ y) {
  __shared__ float sq[Cc], sk[Cc], sa[Cc], sv[DVv], sred[DVv];
  int blk = blockIdx.x;
  int b = blk / Hh, h = blk % Hh;
  int tid = threadIdx.x;
  int dv = tid & 127, chalf = tid >> 7;
  int cb = chalf * 32;

  float S[32];
#pragma unroll
  for (int i = 0; i < 32; ++i) S[i] = 0.f;

  const float* qb = q + (((size_t)b * Hh + h) * Tt) * Cc;
  const float* ab = alpha + (((size_t)b * Hh + h) * Tt) * Cc;
  const float* vb = v + (((size_t)b * Hh + h) * Tt) * DVv;
  const float* kb = kn + ((size_t)b * Tt) * Cc;
  __bf16* yb = y + ((size_t)b * Tt) * (Hh * DVv) + h * DVv;

  for (int t = 0; t < Tt; ++t) {
    // 320 f32 operands per step -> async global->LDS broadcast.
    for (int i = tid; i < 320; i += 256) {
      if (i < 64)
        async_load_b32(lds_off(&sq[i]), qb + (size_t)t * Cc + i);
      else if (i < 128)
        async_load_b32(lds_off(&sk[i - 64]), kb + (size_t)t * Cc + (i - 64));
      else if (i < 192)
        async_load_b32(lds_off(&sa[i - 128]), ab + (size_t)t * Cc + (i - 128));
      else
        async_load_b32(lds_off(&sv[i - 192]), vb + (size_t)t * DVv + (i - 192));
    }
    async_wait0();
    __syncthreads();

    float vv = sv[dv];
    float po = 0.f;
#pragma unroll
    for (int i = 0; i < 32; ++i) {
      int c = cb + i;
      float sx = fmaf(sa[c], S[i], sk[c] * vv);
      S[i] = sx;
      po = fmaf(sq[c], sx, po);
    }
    if (chalf) sred[dv] = po;
    __syncthreads();
    if (!chalf) yb[(size_t)t * (Hh * DVv) + dv] = (__bf16)(po + sred[dv]);
    // next iteration's first barrier orders these reads vs. future writes
  }
}

// ---------------------------------------------------------------------------
// Kernel 5: output projection out = y @ Wo^T + x (residual). WMMA, K=1024.
// Tile staged via async b128 copies; A fragment reused across 8 N-tiles;
// B-fragments preloaded as a clause before the WMMA chain.
// ---------------------------------------------------------------------------

__global__ void __launch_bounds__(256) k_out(const __bf16* __restrict__ y,
                                             const __bf16* __restrict__ woP,
                                             const float* __restrict__ x,
                                             float* __restrict__ out) {
  __shared__ __attribute__((aligned(16))) __bf16 sy[16 * Dd];
  int tile = blockIdx.x, tid = threadIdx.x;

  // 16x1024 bf16 = 32 KB = 2048 x b128.
  const __uint128_t* src = (const __uint128_t*)(y + (size_t)tile * 16 * Dd);
  unsigned sb = lds_off(sy);
  for (int i = tid; i < 2048; i += 256) async_load_b128(sb + i * 16, src + i);
  async_wait0();
  __syncthreads();

  int wave = __builtin_amdgcn_readfirstlane(tid >> 5);
  int lane = tid & 31;
  v8f zero = {};
  v8f acc[8];
#pragma unroll
  for (int j = 0; j < 8; ++j) acc[j] = zero;

  for (int ks = 0; ks < 32; ++ks) {
    v16bf a = load_a_lds(sy, Dd, ks * 32, lane);
    if (ks + 1 < 32)
      __builtin_prefetch(woP + (size_t)(wave * 32 + ks + 1) * 512, 0, 1);
    v16bf bfrag[8];
#pragma unroll
    for (int j = 0; j < 8; ++j)
      bfrag[j] = load_b_panel(woP, (wave + 8 * j) * 32 + ks, lane);
#pragma unroll
    for (int j = 0; j < 8; ++j) acc[j] = wmma_bf16(a, bfrag[j], acc[j]);
  }

  for (int j = 0; j < 8; ++j) {
    int nt = wave + 8 * j;
    int n = nt * 16 + (lane & 15);
    for (int g = 0; g < 8; ++g) {
      int m = g + ((lane < 16) ? 0 : 8);
      size_t bt = (size_t)tile * 16 + m;
      out[bt * Dd + n] = acc[j][g] + x[bt * Dd + n];
    }
  }
}

// ---------------------------------------------------------------------------

extern "C" void kernel_launch(void* const* d_in, const int* in_sizes, int n_in,
                              void* d_out, int out_size, void* d_ws, size_t ws_size,
                              hipStream_t stream) {
  const float* x = (const float*)d_in[0];
  const float* Wq = (const float*)d_in[1];
  const float* Wk = (const float*)d_in[2];
  const float* Wv = (const float*)d_in[3];
  const float* Qc = (const float*)d_in[4];
  const float* Vc = (const float*)d_in[5];
  const float* Wa_w = (const float*)d_in[6];
  const float* Wa_b = (const float*)d_in[7];
  const float* Wo = (const float*)d_in[8];
  const float* gamma = (const float*)d_in[9];
  const float* beta = (const float*)d_in[10];
  float* out = (float*)d_out;
  (void)in_sizes; (void)n_in; (void)out_size; (void)ws_size;

  char* ws = (char*)d_ws;
  size_t off = 0;
  auto alloc = [&](size_t bytes) {
    size_t o = off;
    off += (bytes + 255) & ~(size_t)255;
    return o;
  };
  size_t o_comb  = alloc((size_t)NPROJ * 1024 * 4);      // combined proj weight f32
  size_t o_combP = alloc((size_t)NPROJ_PAD * 1024 * 2);  // padded bf16 panels
  size_t o_qcP   = alloc((size_t)Hh * Cc * Cc * 2);      // Qc panels
  size_t o_vcP   = alloc((size_t)Hh * Cc * DVv * 2);     // Vc panels
  size_t o_woP   = alloc((size_t)Dd * Dd * 2);           // Wo panels
  size_t o_qlat  = alloc((size_t)BT * Cc * 2);           // bf16
  size_t o_vlat  = alloc((size_t)BT * Cc * 2);           // bf16
  size_t o_kn    = alloc((size_t)BT * Cc * 4);           // f32 (normalized in place)
  size_t o_alpha = alloc((size_t)Bb * Hh * Tt * Cc * 4);
  size_t o_q     = alloc((size_t)Bb * Hh * Tt * Cc * 4);
  size_t o_v     = alloc((size_t)Bb * Hh * Tt * DVv * 4);
  size_t o_y     = alloc((size_t)BT * Hh * DVv * 2);     // scan output, bf16

  float*  comb  = (float*)(ws + o_comb);
  __bf16* combP = (__bf16*)(ws + o_combP);
  __bf16* qcP   = (__bf16*)(ws + o_qcP);
  __bf16* vcP   = (__bf16*)(ws + o_vcP);
  __bf16* woP   = (__bf16*)(ws + o_woP);
  __bf16* qlat  = (__bf16*)(ws + o_qlat);
  __bf16* vlat  = (__bf16*)(ws + o_vlat);
  float*  knb   = (float*)(ws + o_kn);
  float*  alp   = (float*)(ws + o_alpha);
  float*  qbuf  = (float*)(ws + o_q);
  float*  vbuf  = (float*)(ws + o_v);
  __bf16* ybuf  = (__bf16*)(ws + o_y);

  // Assemble combined [Wq;Wk;Wv;Wa] (704 x 1024) then panelize all weights.
  hipMemcpyAsync(comb, Wq, (size_t)64 * 1024 * 4, hipMemcpyDeviceToDevice, stream);
  hipMemcpyAsync(comb + 64 * 1024, Wk, (size_t)64 * 1024 * 4, hipMemcpyDeviceToDevice, stream);
  hipMemcpyAsync(comb + 128 * 1024, Wv, (size_t)64 * 1024 * 4, hipMemcpyDeviceToDevice, stream);
  hipMemcpyAsync(comb + 192 * 1024, Wa_w, (size_t)512 * 1024 * 4, hipMemcpyDeviceToDevice, stream);

  k_panel_nk<<<48 * 32, 32, 0, stream>>>(comb, combP, NPROJ, 1024);
  for (int h = 0; h < Hh; ++h) {
    k_panel_kn<<<4 * 2, 32, 0, stream>>>(Qc + (size_t)h * Cc * Cc,
                                         qcP + (size_t)h * 4 * 2 * 512, Cc, Cc);
    k_panel_kn<<<8 * 2, 32, 0, stream>>>(Vc + (size_t)h * Cc * DVv,
                                         vcP + (size_t)h * 8 * 2 * 512, DVv, Cc);
  }
  k_panel_nk<<<64 * 32, 32, 0, stream>>>(Wo, woP, Dd, Dd);

  k_ln_proj<<<BT / 16, 256, 0, stream>>>(x, gamma, beta, Wa_b, combP, qlat, vlat,
                                         knb, alp);
  k_knorm<<<BT, 64, 0, stream>>>(knb);
  k_rotate<<<dim3(BT / 16, Hh), 32, 0, stream>>>(qlat, vlat, qcP, vcP, qbuf, vbuf);
  k_scan<<<Bb * Hh, 256, 0, stream>>>(qbuf, knb, vbuf, alp, ybuf);
  k_out<<<BT / 16, 256, 0, stream>>>(ybuf, woP, x, out);
}
